// SeHGNN_NA_36679020708085
// MI455X (gfx1250) — matrix operand order repaired and checked by reading, not compiled, not run on hardware
//
#include <hip/hip_runtime.h>
#include <math.h>

// ---------------- problem constants ----------------
#define KB   32768
#define KE   524288
#define KNP  100000
#define KNA  60000
#define KNV  8000
#define KDP  512
#define KDA  256
#define KDV  128
#define KF   256
#define KHID 256
#define KGH  8
#define KGD  32
#define KC   4
#define KNC  349
#define KNCP 352   // padded Wout leading dim

typedef __attribute__((ext_vector_type(16))) _Float16 v16h;
typedef __attribute__((ext_vector_type(8)))  float    v8f;
typedef __attribute__((ext_vector_type(8)))  _Float16 h8;
typedef __attribute__((ext_vector_type(2)))  __fp16   f16x2;   // cvt_pkrtz result type
typedef __attribute__((ext_vector_type(4)))  __fp16   f16x4;

// ---------------- WMMA GEMM: C[M,N] = A[M,K] @ B[K,N] (+bias, +prelu) ----------------
// fp32 in/out, f16 compute via v_wmma_f32_16x16x32_f16, fp32 accumulate.
// Block tile 64x128, K-step 32, 8 waves (wave32) in 2x4 grid, each wave 32x32 (2x2 WMMA tiles).
// A staged row-major [64][40] halves; B staged TRANSPOSED [128][40] halves so fragment
// loads are contiguous ds_load_b128. Requires lda/ldb multiples of 4 and 16B-aligned bases.
// FULLN=true: this launch covers only column blocks with bn+BN <= N (no N guards at all).
#define BM 64
#define BN 128
#define BK 32
#define LDSW 40   // half stride per row: 80 bytes -> bank step 20, conflict-free b128 reads

template <bool FULLN>
__global__ __launch_bounds__(256)
void sehgnn_wmma_gemm(const float* __restrict__ A, int lda,
                      const float* __restrict__ Bm, int ldb,
                      float* __restrict__ Cm, int ldc,
                      int M, int N, int K, int bn0,
                      const float* __restrict__ bias,
                      const float* __restrict__ prelu) {
  __shared__ _Float16 As[BM][LDSW];
  __shared__ _Float16 Bs2[BN][LDSW];   // [col][k]

  const int bm = blockIdx.y * BM;
  const int bn = blockIdx.x * BN + bn0;
  const int tid = threadIdx.x;
  const int wave = tid >> 5;
  const int lane = tid & 31;
  const int wm = wave >> 2;   // 0..1
  const int wn = wave & 3;    // 0..3
  const int half = lane >> 4; // 0..1
  const int l16  = lane & 15;

  v8f acc[2][2];
#pragma unroll
  for (int i = 0; i < 2; i++)
#pragma unroll
    for (int j = 0; j < 2; j++)
#pragma unroll
      for (int r = 0; r < 8; r++) acc[i][j][r] = 0.0f;

  // staging coordinates (fixed per thread)
  const int ar0 = (tid) >> 3;            // A: 512 quads: row, col-quad
  const int ac0 = (tid) & 7;
  const int ar1 = (tid + 256) >> 3;
  const int ac1 = (tid + 256) & 7;
  const int bk0 = (tid) >> 5;            // B: 512 chunks: k-pair, col-quad
  const int bc0 = (tid) & 31;
  const int bk1 = (tid + 256) >> 5;
  const int bc1 = (tid + 256) & 31;

  for (int k0 = 0; k0 < K; k0 += BK) {
    // ---- stage A tile (unguarded b128 loads; row clamped — extra rows never stored) ----
#pragma unroll
    for (int i = 0; i < 2; i++) {
      int r  = i ? ar1 : ar0;
      int cq = i ? ac1 : ac0;
      int gr = bm + r; if (gr > M - 1) gr = M - 1;
      float4 v = *(const float4*)(A + (size_t)gr * lda + k0 + cq * 4);
      f16x2 p01 = __builtin_amdgcn_cvt_pkrtz(v.x, v.y);
      f16x2 p23 = __builtin_amdgcn_cvt_pkrtz(v.z, v.w);
      f16x4 hv = __builtin_shufflevector(p01, p23, 0, 1, 2, 3);
      *(f16x4*)&As[r][cq * 4] = hv;   // ds_store_b64
    }
    // ---- stage B tile transposed (k-pairs packed into b32 stores) ----
#pragma unroll
    for (int i = 0; i < 2; i++) {
      int kp = i ? bk1 : bk0;
      int cq = i ? bc1 : bc0;
      int gq = bn + cq * 4;
      int kr = k0 + 2 * kp;
      float e00, e01, e02, e03, e10, e11, e12, e13;
      if (FULLN || gq + 3 < N) {
        float4 v0 = *(const float4*)(Bm + (size_t)kr * ldb + gq);
        float4 v1 = *(const float4*)(Bm + (size_t)(kr + 1) * ldb + gq);
        e00 = v0.x; e01 = v0.y; e02 = v0.z; e03 = v0.w;
        e10 = v1.x; e11 = v1.y; e12 = v1.z; e13 = v1.w;
      } else {
        const float* r0 = Bm + (size_t)kr * ldb;
        const float* r1 = Bm + (size_t)(kr + 1) * ldb;
        e00 = (gq + 0 < N) ? r0[gq + 0] : 0.0f;
        e01 = (gq + 1 < N) ? r0[gq + 1] : 0.0f;
        e02 = (gq + 2 < N) ? r0[gq + 2] : 0.0f;
        e03 = (gq + 3 < N) ? r0[gq + 3] : 0.0f;
        e10 = (gq + 0 < N) ? r1[gq + 0] : 0.0f;
        e11 = (gq + 1 < N) ? r1[gq + 1] : 0.0f;
        e12 = (gq + 2 < N) ? r1[gq + 2] : 0.0f;
        e13 = (gq + 3 < N) ? r1[gq + 3] : 0.0f;
      }
      *(f16x2*)&Bs2[cq * 4 + 0][2 * kp] = __builtin_amdgcn_cvt_pkrtz(e00, e10);
      *(f16x2*)&Bs2[cq * 4 + 1][2 * kp] = __builtin_amdgcn_cvt_pkrtz(e01, e11);
      *(f16x2*)&Bs2[cq * 4 + 2][2 * kp] = __builtin_amdgcn_cvt_pkrtz(e02, e12);
      *(f16x2*)&Bs2[cq * 4 + 3][2 * kp] = __builtin_amdgcn_cvt_pkrtz(e03, e13);
    }
    // prefetch next K tile while this one computes
    if (k0 + 2 * BK <= K) {
      int gr = bm + (tid & 63); if (gr > M - 1) gr = M - 1;
      __builtin_prefetch(A + (size_t)gr * lda + k0 + BK, 0, 1);
      __builtin_prefetch(Bm + (size_t)(k0 + BK + (tid >> 3)) * ldb + bn, 0, 1);
    }
    __syncthreads();

    // ---- fragments: contiguous 2x ds_load_b128 each ----
    v16h af[2], bf[2];
#pragma unroll
    for (int ti = 0; ti < 2; ti++) {
      int row = wm * 32 + ti * 16 + l16;
      h8 a0 = *(const h8*)&As[row][half * 8];
      h8 a1 = *(const h8*)&As[row][16 + half * 8];
      af[ti] = __builtin_shufflevector(a0, a1, 0, 1, 2, 3, 4, 5, 6, 7,
                                       8, 9, 10, 11, 12, 13, 14, 15);
    }
#pragma unroll
    for (int tj = 0; tj < 2; tj++) {
      int col = wn * 32 + tj * 16 + l16;
      h8 b0 = *(const h8*)&Bs2[col][half * 8];
      h8 b1 = *(const h8*)&Bs2[col][16 + half * 8];
      bf[tj] = __builtin_shufflevector(b0, b1, 0, 1, 2, 3, 4, 5, 6, 7,
                                       8, 9, 10, 11, 12, 13, 14, 15);
    }

#pragma unroll
    for (int ti = 0; ti < 2; ti++)
#pragma unroll
      for (int tj = 0; tj < 2; tj++)
        acc[ti][tj] = __builtin_amdgcn_wmma_f32_16x16x32_f16(
            false, af[ti], false, bf[tj], (short)0, acc[ti][tj], false, false);
    __syncthreads();
  }

  // Epilogue: C/D layout: col = lane%16, VGPR r -> row = r + (lane/16)*8
#pragma unroll
  for (int ti = 0; ti < 2; ti++)
#pragma unroll
    for (int tj = 0; tj < 2; tj++) {
      int rbase = bm + wm * 32 + ti * 16 + half * 8;
      int col = bn + wn * 32 + tj * 16 + l16;
      if (FULLN || col < N) {
        float bv = bias ? bias[col] : 0.0f;
#pragma unroll
        for (int r = 0; r < 8; r++) {
          int row = rbase + r;
          if (row < M) {
            float v = acc[ti][tj][r] + bv;
            if (prelu) { float a = prelu[0]; v = (v >= 0.0f) ? v : a * v; }
            Cm[(size_t)row * ldc + col] = v;
          }
        }
      }
    }
}

// ---------------- helpers ----------------
__device__ __forceinline__ unsigned f32_ord(float f) {
  unsigned u = __float_as_uint(f);
  return (u & 0x80000000u) ? ~u : (u | 0x80000000u);
}
__device__ __forceinline__ float ord_f32(unsigned u) {
  unsigned v = (u & 0x80000000u) ? (u & 0x7fffffffu) : ~u;
  return __uint_as_float(v);
}

__global__ void sehgnn_fill_f32(float* p, float v, int n) {
  int i = blockIdx.x * 256 + threadIdx.x;
  if (i < n) p[i] = v;
}
__global__ void sehgnn_fill_u32(unsigned* p, unsigned v, int n) {
  int i = blockIdx.x * 256 + threadIdx.x;
  if (i < n) p[i] = v;
}

__global__ void sehgnn_gather_rows(const float* __restrict__ Mt, const int* __restrict__ idxv,
                                   float* __restrict__ out) {
  int i = blockIdx.x * 256 + threadIdx.x;   // KB*KF/4 quads
  int b = i >> 6, j = (i & 63) * 4;
  ((float4*)out)[i] = *(const float4*)(Mt + (size_t)idxv[b] * KF + j);
}

__global__ void sehgnn_pad_wout(const float* __restrict__ src, float* __restrict__ dst) {
  int i = blockIdx.x * 256 + threadIdx.x;   // KHID*KNCP exact
  int r = i / KNCP, c = i - r * KNCP;
  dst[i] = (c < KNC) ? src[(size_t)r * KNC + c] : 0.0f;
}

// al[n,h] = sum_d X[n, h*32+d] * a[h,d]
__global__ void sehgnn_compute_al(const float* __restrict__ X, const float* __restrict__ avec,
                                  float* __restrict__ al, int n) {
  int idx = blockIdx.x * 256 + threadIdx.x;
  if (idx >= n * KGH) return;
  int node = idx >> 3, h = idx & 7;
  const float* xp = X + (size_t)node * KHID + h * KGD;
  const float* ap = avec + h * KGD;
  float s = 0.0f;
#pragma unroll
  for (int d = 0; d < KGD; d++) s += xp[d] * ap[d];
  al[idx] = s;
}

__global__ void sehgnn_edge_alpha(const float* __restrict__ al_s, const float* __restrict__ al_d,
                                  const int* __restrict__ src, const int* __restrict__ dst,
                                  float* __restrict__ edgew, unsigned* __restrict__ maxb) {
  int idx = blockIdx.x * 256 + threadIdx.x;   // KE*KGH exact
  int e = idx >> 3, h = idx & 7;
  int s = src[e], d = dst[e];
  float a = al_s[(size_t)s * KGH + h] + al_d[(size_t)d * KGH + h];
  a = (a >= 0.0f) ? a : 0.2f * a;            // leaky_relu(0.2)
  edgew[idx] = a;
  atomicMax(&maxb[(size_t)d * KGH + h], f32_ord(a));
}

__global__ void sehgnn_edge_exp(float* __restrict__ edgew, const unsigned* __restrict__ maxb,
                                const int* __restrict__ dst, float* __restrict__ denom) {
  int idx = blockIdx.x * 256 + threadIdx.x;   // KE*KGH exact
  int e = idx >> 3, h = idx & 7;
  int d = dst[e];
  float m = ord_f32(maxb[(size_t)d * KGH + h]);
  float ev = __expf(edgew[idx] - m);
  edgew[idx] = ev;
  atomicAdd(&denom[(size_t)d * KGH + h], ev);
}

// one wave per edge; each lane scatters 8 contiguous channels (all within one head)
__global__ void sehgnn_edge_scatter(const float* __restrict__ xs, const int* __restrict__ src,
                                    const int* __restrict__ dst, const float* __restrict__ edgew,
                                    const float* __restrict__ denom, float* __restrict__ out) {
  long long t = (long long)blockIdx.x * 256 + threadIdx.x;  // KE*32 exact
  int e = (int)(t >> 5);
  int lane = (int)(t & 31);
  int h = lane >> 2;
  int s = src[e], d = dst[e];
  float w = edgew[(size_t)e * KGH + h] / (denom[(size_t)d * KGH + h] + 1e-16f);
  const float4* xp = (const float4*)(xs + (size_t)s * KHID + lane * 8);
  float4 v0 = xp[0], v1 = xp[1];
  float* op = out + (size_t)d * KHID + lane * 8;
  atomicAdd(&op[0], v0.x * w); atomicAdd(&op[1], v0.y * w);
  atomicAdd(&op[2], v0.z * w); atomicAdd(&op[3], v0.w * w);
  atomicAdd(&op[4], v1.x * w); atomicAdd(&op[5], v1.y * w);
  atomicAdd(&op[6], v1.z * w); atomicAdd(&op[7], v1.w * w);
}

__global__ void sehgnn_stack_x0(const float* __restrict__ a0, const float* __restrict__ a1,
                                const float* __restrict__ a2, const float* __restrict__ tgt,
                                const float* __restrict__ b0, const float* __restrict__ b1,
                                const float* __restrict__ b2, float* __restrict__ x0) {
  int idx = blockIdx.x * 256 + threadIdx.x;  // KB*1024 exact
  int b = idx >> 10, r = idx & 1023, c = r >> 8, j = r & 255;
  size_t bi = (size_t)b * KHID + j;
  float v;
  if (c == 0)      v = a0[bi] + b0[j];
  else if (c == 1) v = a1[bi] + b1[j];
  else if (c == 2) v = a2[bi] + b2[j];
  else             v = tgt[bi];
  x0[idx] = v;
}

// LayerNorm over [C*HID]=1024 per row, then affine (w,b per element), then PReLU(scalar)
__global__ void sehgnn_ln_prelu(float* __restrict__ X, const float* __restrict__ w,
                                const float* __restrict__ bb, const float* __restrict__ p) {
  __shared__ float s1[256], s2[256];
  int b = blockIdx.x, t = threadIdx.x;
  size_t base = (size_t)b * (KC * KHID);
  float sum = 0.0f, sq = 0.0f;
#pragma unroll
  for (int i = 0; i < 4; i++) { float v = X[base + t + i * 256]; sum += v; sq += v * v; }
  s1[t] = sum; s2[t] = sq; __syncthreads();
  for (int st = 128; st > 0; st >>= 1) {
    if (t < st) { s1[t] += s1[t + st]; s2[t] += s2[t + st]; }
    __syncthreads();
  }
  float mean = s1[0] * (1.0f / 1024.0f);
  float var  = s2[0] * (1.0f / 1024.0f) - mean * mean;
  float rstd = rsqrtf(var + 1e-5f);
  float a = p[0];
#pragma unroll
  for (int i = 0; i < 4; i++) {
    int j = t + i * 256;
    float v = (X[base + j] - mean) * rstd * w[j] + bb[j];
    X[base + j] = (v >= 0.0f) ? v : a * v;
  }
}

// per-row 4x4 single-head attention: Y = gamma * softmax(F G^T / 8) H + Xin
__global__ void sehgnn_attention(const float* __restrict__ Fm, const float* __restrict__ Gm,
                                 const float* __restrict__ Hm, const float* __restrict__ Xin,
                                 const float* __restrict__ gamma, float* __restrict__ Y) {
  __shared__ float fsh[256], gsh[256], hsh[1024], betash[16];
  int b = blockIdx.x, t = threadIdx.x;
  fsh[t] = Fm[(size_t)b * 256 + t];
  gsh[t] = Gm[(size_t)b * 256 + t];
#pragma unroll
  for (int i = 0; i < 4; i++) hsh[t + i * 256] = Hm[(size_t)b * 1024 + t + i * 256];
  __syncthreads();
  if (t < 16) {
    int m = t >> 2, n = t & 3;
    float s = 0.0f;
#pragma unroll
    for (int k2 = 0; k2 < 64; k2++) s += fsh[m * 64 + k2] * gsh[n * 64 + k2];
    betash[t] = s * 0.125f;
  }
  __syncthreads();
  if (t < 4) {
    float mx = betash[t * 4];
    for (int n = 1; n < 4; n++) mx = fmaxf(mx, betash[t * 4 + n]);
    float e0[4], ss = 0.0f;
    for (int n = 0; n < 4; n++) { e0[n] = __expf(betash[t * 4 + n] - mx); ss += e0[n]; }
    float inv = 1.0f / ss;
    for (int n = 0; n < 4; n++) betash[t * 4 + n] = e0[n] * inv;
  }
  __syncthreads();
  float g0 = gamma[0];
#pragma unroll
  for (int i = 0; i < 4; i++) {
    int j = t + i * 256, m = j >> 8, cc = j & 255;
    float v = 0.0f;
#pragma unroll
    for (int n = 0; n < 4; n++) v += betash[m * 4 + n] * hsh[n * 256 + cc];
    Y[(size_t)b * 1024 + j] = g0 * v + Xin[(size_t)b * 1024 + j];
  }
}

// XF[b, j*4 + c] = Y[b, c*256 + j]   (transpose (0,2,1) + reshape)
__global__ void sehgnn_transpose_cx(const float* __restrict__ Y, float* __restrict__ XF) {
  int idx = blockIdx.x * 256 + threadIdx.x;  // KB*1024 exact
  int b = idx >> 10, r = idx & 1023, c = r >> 8, j = r & 255;
  XF[(size_t)b * 1024 + j * 4 + c] = Y[idx];
}

__global__ void sehgnn_bn_stats(const float* __restrict__ X, int ncols,
                                float* __restrict__ mean, float* __restrict__ rstd) {
  __shared__ float s1[256], s2[256];
  int col = blockIdx.x, t = threadIdx.x;
  float sum = 0.0f, sq = 0.0f;
  for (int r = t; r < KB; r += 256) {
    float v = X[(size_t)r * ncols + col];
    sum += v; sq += v * v;
  }
  s1[t] = sum; s2[t] = sq; __syncthreads();
  for (int st = 128; st > 0; st >>= 1) {
    if (t < st) { s1[t] += s1[t + st]; s2[t] += s2[t + st]; }
    __syncthreads();
  }
  if (t == 0) {
    float m = s1[0] * (1.0f / KB);
    float var = s2[0] * (1.0f / KB) - m * m;
    mean[col] = m;
    rstd[col] = rsqrtf(var + 1e-5f);
  }
}

__global__ void sehgnn_bn_apply(const float* __restrict__ X, int ncols,
                                const float* __restrict__ mean, const float* __restrict__ rstd,
                                const float* __restrict__ p, float* __restrict__ Y,
                                long long total) {
  long long idx = (long long)blockIdx.x * 256 + threadIdx.x;
  if (idx >= total) return;
  int col = (int)(idx % ncols);
  float v = (X[idx] - mean[col]) * rstd[col];
  if (p) { float a = p[0]; v = (v >= 0.0f) ? v : a * v; }
  Y[idx] = v;
}

// ---------------- host side ----------------
static inline void launch_gemm(const float* A, int lda, const float* Bm, int ldb,
                               float* Cm, int ldc, int M, int N, int K,
                               const float* bias, const float* prelu, hipStream_t s) {
  int gy = (M + BM - 1) / BM;
  int nfull = N / BN;                       // column blocks entirely inside N
  if (nfull > 0)
    sehgnn_wmma_gemm<true><<<dim3(nfull, gy), 256, 0, s>>>(
        A, lda, Bm, ldb, Cm, ldc, M, N, K, 0, bias, prelu);
  if (N - nfull * BN > 0)
    sehgnn_wmma_gemm<false><<<dim3(1, gy), 256, 0, s>>>(
        A, lda, Bm, ldb, Cm, ldc, M, N, K, nfull * BN, bias, prelu);
}

extern "C" void kernel_launch(void* const* d_in, const int* in_sizes, int n_in,
                              void* d_out, int out_size, void* d_ws, size_t ws_size,
                              hipStream_t stream) {
  (void)in_sizes; (void)n_in; (void)out_size; (void)ws_size;

  const float* feat_P  = (const float*)d_in[0];
  const float* feat_A  = (const float*)d_in[1];
  const float* feat_V  = (const float*)d_in[2];
  const int*   batch   = (const int*)d_in[3];
  const int*   src_PA  = (const int*)d_in[4];
  const int*   dst_PA  = (const int*)d_in[5];
  const int*   src_PP  = (const int*)d_in[6];
  const int*   dst_PP  = (const int*)d_in[7];
  const int*   src_PV  = (const int*)d_in[8];
  const int*   dst_PV  = (const int*)d_in[9];
  const float* emb_P   = (const float*)d_in[10];
  const float* emb_A   = (const float*)d_in[11];
  const float* emb_V   = (const float*)d_in[12];
  const float* Wg_PA   = (const float*)d_in[13];
  const float* asrc_PA = (const float*)d_in[14];
  const float* adst_PA = (const float*)d_in[15];
  const float* bg_PA   = (const float*)d_in[16];
  const float* Wg_PP   = (const float*)d_in[17];
  const float* asrc_PP = (const float*)d_in[18];
  const float* adst_PP = (const float*)d_in[19];
  const float* bg_PP   = (const float*)d_in[20];
  const float* Wg_PV   = (const float*)d_in[21];
  const float* asrc_PV = (const float*)d_in[22];
  const float* adst_PV = (const float*)d_in[23];
  const float* bg_PV   = (const float*)d_in[24];
  const float* W1      = (const float*)d_in[25];
  const float* b1      = (const float*)d_in[26];
  const float* ln1_w   = (const float*)d_in[27];
  const float* ln1_b   = (const float*)d_in[28];
  const float* p1      = (const float*)d_in[29];
  const float* W2      = (const float*)d_in[30];
  const float* b2      = (const float*)d_in[31];
  const float* ln2_w   = (const float*)d_in[32];
  const float* ln2_b   = (const float*)d_in[33];
  const float* p2      = (const float*)d_in[34];
  const float* Wq      = (const float*)d_in[35];
  const float* bq      = (const float*)d_in[36];
  const float* Wk      = (const float*)d_in[37];
  const float* bk      = (const float*)d_in[38];
  const float* Wv      = (const float*)d_in[39];
  const float* bv      = (const float*)d_in[40];
  const float* gamma   = (const float*)d_in[41];
  const float* Wf      = (const float*)d_in[42];
  const float* bf      = (const float*)d_in[43];
  const float* p3      = (const float*)d_in[44];
  const float* Wl1     = (const float*)d_in[45];
  const float* bl1     = (const float*)d_in[46];
  const float* p4      = (const float*)d_in[47];
  const float* Wout    = (const float*)d_in[48];

  float* W = (float*)d_ws;
  size_t off = 0;
  auto alloc = [&](size_t nfloats) { size_t o = off; off += (nfloats + 63) & ~(size_t)63; return o; };

  // persistent buffers
  size_t o_mP   = alloc((size_t)KNP * KF);
  size_t o_tgt  = alloc((size_t)KB * KHID);
  size_t o_a0   = alloc((size_t)KB * KHID);
  size_t o_a1   = alloc((size_t)KB * KHID);
  size_t o_a2   = alloc((size_t)KB * KHID);
  size_t o_wout = alloc((size_t)KHID * KNCP);   // padded Wout
  // phase-2 big buffers (GAT scratch aliases this region in phase 1)
  size_t o_x0  = alloc((size_t)KB * KC * KHID);
  size_t o_y   = alloc((size_t)KB * KC * KHID);
  size_t o_h   = alloc((size_t)KB * KC * KHID);
  size_t o_f   = alloc((size_t)KB * KC * 64);
  size_t o_g   = alloc((size_t)KB * KC * 64);
  size_t o_m1  = alloc(KHID);
  size_t o_r1  = alloc(KHID);
  size_t o_m2  = alloc(KNCP);
  size_t o_r2  = alloc(KNCP);

  // GAT scratch carved out of [o_x0 ...] region (free during phase 1)
  size_t goff = o_x0;
  auto galloc = [&](size_t nfloats) { size_t o = goff; goff += (nfloats + 63) & ~(size_t)63; return o; };
  size_t o_mX   = galloc((size_t)KNA * KF);        // largest non-P projection
  size_t o_xs   = galloc((size_t)KNP * KHID);      // largest xs (PP)
  size_t o_xd   = galloc((size_t)KB * KHID);
  size_t o_als  = galloc((size_t)KNP * KGH);
  size_t o_ald  = galloc((size_t)KB * KGH);
  size_t o_ew   = galloc((size_t)KE * KGH);
  size_t o_maxb = galloc((size_t)KB * KGH);
  size_t o_den  = galloc((size_t)KB * KGH);

  float* mP  = W + o_mP;
  float* tgt = W + o_tgt;
  float* xs  = W + o_xs;
  float* xd  = W + o_xd;
  float* als = W + o_als;
  float* ald = W + o_ald;
  float* ew  = W + o_ew;
  unsigned* maxb = (unsigned*)(W + o_maxb);
  float* den = W + o_den;

  const int nBH   = KB * KGH;
  const int nEH   = KE * KGH;
  const int nBHID = KB * KHID;

  // ---- Phase 1: projections ----
  launch_gemm(feat_P, KDP, emb_P, KF, mP, KF, KNP, KF, KDP, nullptr, nullptr, stream);
  sehgnn_gather_rows<<<(KB * KF / 4) / 256, 256, 0, stream>>>(mP, batch, tgt);
  sehgnn_pad_wout<<<(KHID * KNCP) / 256, 256, 0, stream>>>(Wout, W + o_wout);

  // ---- GAT per edge type ----
  auto run_gat = [&](const float* xsrc, int NX, const float* Wg, const float* as_, const float* ad_,
                     const int* src, const int* dst, float* aout) {
    launch_gemm(xsrc, KF, Wg, KHID, xs, KHID, NX, KHID, KF, nullptr, nullptr, stream);
    launch_gemm(tgt,  KF, Wg, KHID, xd, KHID, KB, KHID, KF, nullptr, nullptr, stream);
    sehgnn_compute_al<<<(NX * KGH + 255) / 256, 256, 0, stream>>>(xs, as_, als, NX);
    sehgnn_compute_al<<<(KB * KGH + 255) / 256, 256, 0, stream>>>(xd, ad_, ald, KB);
    sehgnn_fill_u32<<<(nBH + 255) / 256, 256, 0, stream>>>(maxb, 0x007FFFFFu, nBH); // ord(-inf)
    sehgnn_fill_f32<<<(nBH + 255) / 256, 256, 0, stream>>>(den, 0.0f, nBH);
    sehgnn_edge_alpha<<<nEH / 256, 256, 0, stream>>>(als, ald, src, dst, ew, maxb);
    sehgnn_edge_exp<<<nEH / 256, 256, 0, stream>>>(ew, maxb, dst, den);
    sehgnn_fill_f32<<<(nBHID + 255) / 256, 256, 0, stream>>>(aout, 0.0f, nBHID);
    sehgnn_edge_scatter<<<(KE * 32) / 256, 256, 0, stream>>>(xs, src, dst, ew, den, aout);
  };

  // PA: project feat_A first
  launch_gemm(feat_A, KDA, emb_A, KF, W + o_mX, KF, KNA, KF, KDA, nullptr, nullptr, stream);
  run_gat(W + o_mX, KNA, Wg_PA, asrc_PA, adst_PA, src_PA, dst_PA, W + o_a0);
  // PP: source nodes are mP
  run_gat(mP, KNP, Wg_PP, asrc_PP, adst_PP, src_PP, dst_PP, W + o_a1);
  // PV
  launch_gemm(feat_V, KDV, emb_V, KF, W + o_mX, KF, KNV, KF, KDV, nullptr, nullptr, stream);
  run_gat(W + o_mX, KNV, Wg_PV, asrc_PV, adst_PV, src_PV, dst_PV, W + o_a2);

  // ---- Phase 2: stack + grouped convs + LN/PReLU ----
  float* x0 = W + o_x0;   // GAT scratch now dead
  float* y  = W + o_y;
  float* hb = W + o_h;
  float* fb = W + o_f;
  float* gb = W + o_g;

  sehgnn_stack_x0<<<(KB * 1024) / 256, 256, 0, stream>>>(W + o_a0, W + o_a1, W + o_a2, tgt,
                                                         bg_PA, bg_PP, bg_PV, x0);
  for (int c = 0; c < KC; c++)
    launch_gemm(x0 + c * KHID, KC * KHID, W1 + (size_t)c * KF * KHID, KHID,
                y + c * KHID, KC * KHID, KB, KHID, KF, b1 + c * KHID, nullptr, stream);
  sehgnn_ln_prelu<<<KB, 256, 0, stream>>>(y, ln1_w, ln1_b, p1);
  for (int c = 0; c < KC; c++)
    launch_gemm(y + c * KHID, KC * KHID, W2 + (size_t)c * KHID * KHID, KHID,
                x0 + c * KHID, KC * KHID, KB, KHID, KHID, b2 + c * KHID, nullptr, stream);
  sehgnn_ln_prelu<<<KB, 256, 0, stream>>>(x0, ln2_w, ln2_b, p2);

  // ---- attention over C=4 channels ----
  launch_gemm(x0, KHID, Wq, 64, fb, 64, KB * KC, 64, KHID, bq, nullptr, stream);
  launch_gemm(x0, KHID, Wk, 64, gb, 64, KB * KC, 64, KHID, bk, nullptr, stream);
  launch_gemm(x0, KHID, Wv, KHID, hb, KHID, KB * KC, KHID, KHID, bv, nullptr, stream);
  sehgnn_attention<<<KB, 256, 0, stream>>>(fb, gb, hb, x0, gamma, y);
  sehgnn_transpose_cx<<<(KB * 1024) / 256, 256, 0, stream>>>(y, x0);  // xf in x0

  // ---- head MLP ----
  float* outf = W + o_a0;   // reuse
  float* h1   = W + o_a1;   // reuse
  launch_gemm(x0, KC * KHID, Wf, KHID, outf, KHID, KB, KHID, KC * KHID, bf, p3, stream);
  launch_gemm(outf, KHID, Wl1, KHID, h1, KHID, KB, KHID, KHID, bl1, nullptr, stream);
  sehgnn_bn_stats<<<KHID, 256, 0, stream>>>(h1, KHID, W + o_m1, W + o_r1);
  sehgnn_bn_apply<<<(nBHID + 255) / 256, 256, 0, stream>>>(h1, KHID, W + o_m1, W + o_r1,
                                                           p4, h1, (long long)nBHID);
  float* logits = fb;       // reuse f+g region (contiguous, large enough)
  launch_gemm(h1, KHID, W + o_wout, KNCP, logits, KNC, KB, KNC, KHID, nullptr, nullptr, stream);
  sehgnn_bn_stats<<<KNC, 256, 0, stream>>>(logits, KNC, W + o_m2, W + o_r2);
  long long ntot = (long long)KB * KNC;
  sehgnn_bn_apply<<<(int)((ntot + 255) / 256), 256, 0, stream>>>(logits, KNC, W + o_m2, W + o_r2,
                                                                 nullptr, (float*)d_out, ntot);
}
